// Local_Attention_26895085208068
// MI455X (gfx1250) — matrix operand order repaired and checked by reading, not compiled, not run on hardware
//
#include <hip/hip_runtime.h>
#include <hip/hip_bf16.h>

typedef __attribute__((ext_vector_type(16))) _Float16 v16h;
typedef __attribute__((ext_vector_type(8)))  _Float16 v8h;
typedef __attribute__((ext_vector_type(4)))  _Float16 v4h;
typedef __attribute__((ext_vector_type(8)))  float    v8f;

#define WMMA16(A, B, C) __builtin_amdgcn_wmma_f32_16x16x32_f16( \
    false, (A), false, (B), (short)0, (C), false, false)

#define BB 2
#define TT 2048
#define EE 1024
#define HH 16
#define DD 64
#define WW 128
#define SCALE 0.125f            // 1/sqrt(64)
#define MASK_NEG  (-3.0e38f)
#define MINIT_NEG (-1.0e38f)

// ---------------------------------------------------------------------------
// Fragment loaders (row-major f16 source).
// A-frag 16x32: lane holds row (lane&15); K chunks [hi*8, hi*8+8) and +16.
// B-frag 32x16: lane holds col (lane&15); K chunks [hi*8, hi*8+8) and +16.
// ---------------------------------------------------------------------------
__device__ __forceinline__ v16h frag_a(const _Float16* rowp, int k0, int hi) {
    const _Float16* c = rowp + k0 + hi * 8;
    v8h lo = *(const v8h*)(c);
    v8h hh = *(const v8h*)(c + 16);
    v16h f;
#pragma unroll
    for (int i = 0; i < 8; ++i) { f[i] = lo[i]; f[i + 8] = hh[i]; }
    return f;
}

__device__ __forceinline__ v16h frag_b(const _Float16* colp, int k0, int hi) {
    const _Float16* c = colp + k0 + hi * 8;
    v8h lo = *(const v8h*)(c);
    v8h hh = *(const v8h*)(c + 16);
    v16h f;
#pragma unroll
    for (int i = 0; i < 8; ++i) { f[i] = lo[i]; f[i + 8] = hh[i]; }
    return f;
}

// ---------------------------------------------------------------------------
// fp32 -> f16 convert (vectorized x4)
// ---------------------------------------------------------------------------
__global__ __launch_bounds__(256) void cvt_f32_f16(const float* __restrict__ s,
                                                   _Float16* __restrict__ d, int n) {
    int i = (blockIdx.x * blockDim.x + threadIdx.x) * 4;
    if (i < n) {
        float4 v = *(const float4*)(s + i);
        v4h h = { (_Float16)v.x, (_Float16)v.y, (_Float16)v.z, (_Float16)v.w };
        *(v4h*)(d + i) = h;
    }
}

// ---------------------------------------------------------------------------
// Software-pipelined 16x64-per-wave GEMM core: acc[4] += A[16xK] * B^T tiles.
// Next iteration's fragments are loaded before current WMMAs so vector loads
// overlap matrix math instead of stalling each v_wmma on s_wait_loadcnt 0.
// ---------------------------------------------------------------------------
__device__ __forceinline__ void gemm_core(const _Float16* __restrict__ arow,
                                          const _Float16* __restrict__ b0r,
                                          const _Float16* __restrict__ b1r,
                                          const _Float16* __restrict__ b2r,
                                          const _Float16* __restrict__ b3r,
                                          int hi, v8f acc[4]) {
    v16h a  = frag_a(arow, 0, hi);
    v16h b0 = frag_b(b0r, 0, hi);
    v16h b1 = frag_b(b1r, 0, hi);
    v16h b2 = frag_b(b2r, 0, hi);
    v16h b3 = frag_b(b3r, 0, hi);
    for (int kk = 0; kk < EE - 32; kk += 32) {
        const int kn = kk + 32;
        __builtin_prefetch(arow + kn + 128, 0, 3);
        v16h an  = frag_a(arow, kn, hi);
        v16h bn0 = frag_b(b0r, kn, hi);
        v16h bn1 = frag_b(b1r, kn, hi);
        v16h bn2 = frag_b(b2r, kn, hi);
        v16h bn3 = frag_b(b3r, kn, hi);
        acc[0] = WMMA16(a, b0, acc[0]);
        acc[1] = WMMA16(a, b1, acc[1]);
        acc[2] = WMMA16(a, b2, acc[2]);
        acc[3] = WMMA16(a, b3, acc[3]);
        a = an; b0 = bn0; b1 = bn1; b2 = bn2; b3 = bn3;
    }
    acc[0] = WMMA16(a, b0, acc[0]);
    acc[1] = WMMA16(a, b1, acc[1]);
    acc[2] = WMMA16(a, b2, acc[2]);
    acc[3] = WMMA16(a, b3, acc[3]);
}

// ---------------------------------------------------------------------------
// QKV GEMM: C[4096,3072] = Xh[4096,1024] @ Wqkv^T. Block = 4 waves, each wave
// does a 16x64 C tile. Q scaled by 1/sqrt(D); V stored transposed [B,H,D,T].
// ---------------------------------------------------------------------------
__global__ __launch_bounds__(128) void qkv_gemm(const _Float16* __restrict__ X,
                                                const _Float16* __restrict__ Wq,
                                                _Float16* __restrict__ Q,
                                                _Float16* __restrict__ K,
                                                _Float16* __restrict__ V) {
    const int lane = threadIdx.x & 31;
    const int wave = threadIdx.x >> 5;
    const int hi   = lane >> 4;
    const int lo16 = lane & 15;
    const int m0 = blockIdx.x * 64 + wave * 16;
    const int n0 = blockIdx.y * 64;

    const _Float16* arow = X + (size_t)(m0 + lo16) * EE;
    v8f acc[4] = {{}, {}, {}, {}};
    gemm_core(arow,
              Wq + (size_t)(n0 +  0 + lo16) * EE,
              Wq + (size_t)(n0 + 16 + lo16) * EE,
              Wq + (size_t)(n0 + 32 + lo16) * EE,
              Wq + (size_t)(n0 + 48 + lo16) * EE, hi, acc);

#pragma unroll
    for (int nt = 0; nt < 4; ++nt) {
        int n = n0 + nt * 16 + lo16;
        int seg = n >> 10;          // 0=Q 1=K 2=V
        int nn  = n & 1023;
        int h = nn >> 6, d = nn & 63;
#pragma unroll
        for (int r = 0; r < 8; ++r) {
            int m = m0 + r + 8 * hi;
            int b = m >> 11, t = m & 2047;
            size_t bh = (size_t)(b * HH + h);
            float v = acc[nt][r];
            if (seg == 0)      Q[(bh * TT + t) * DD + d] = (_Float16)(v * SCALE);
            else if (seg == 1) K[(bh * TT + t) * DD + d] = (_Float16)v;
            else               V[(bh * DD + d) * TT + t] = (_Float16)v;
        }
    }
}

// ---------------------------------------------------------------------------
// Sliding-window flash attention. One wave per 16-query tile; <=9 key tiles.
// S = Q K^T via WMMA, online softmax (16-lane butterflies), P staged through
// LDS (C-layout -> A-layout), O += P V via WMMA with Vt [D,T] B-frags.
// ---------------------------------------------------------------------------
__global__ __launch_bounds__(128) void attn_kernel(const _Float16* __restrict__ Q,
                                                   const _Float16* __restrict__ K,
                                                   const _Float16* __restrict__ Vt,
                                                   _Float16* __restrict__ Ao) {
    __shared__ _Float16 pst[4][16 * 32];
    const int lane = threadIdx.x & 31;
    const int wave = threadIdx.x >> 5;
    const int hi   = lane >> 4;
    const int lo16 = lane & 15;
    const int qtile = blockIdx.x * 4 + wave;
    const int bh = blockIdx.y;                  // b*H + h
    const int q0 = qtile * 16;

    const _Float16* Qp = Q  + (size_t)bh * TT * DD;
    const _Float16* Kp = K  + (size_t)bh * TT * DD;
    const _Float16* Vp = Vt + (size_t)bh * DD * TT;

    const _Float16* qrow = Qp + (size_t)(q0 + lo16) * DD;
    v16h aq0 = frag_a(qrow, 0, hi);
    v16h aq1 = frag_a(qrow, 32, hi);

    float mrow[8], lrow[8];
    v8f o[4] = {{}, {}, {}, {}};
#pragma unroll
    for (int r = 0; r < 8; ++r) { mrow[r] = MINIT_NEG; lrow[r] = 0.f; }

    int losub = q0 - (WW - 1);
    if (losub < 0) losub = 0;
    const int t0 = losub >> 4;
    _Float16* st = pst[wave];

    for (int kt2 = t0; kt2 <= qtile; kt2 += 2) {
#pragma unroll
        for (int s = 0; s < 2; ++s) {
            int kt = kt2 + s;
            if (kt <= qtile) {
                const _Float16* krow = Kp + (size_t)(kt * 16 + lo16) * DD;
                v16h bk0 = frag_b(krow, 0, hi);
                v16h bk1 = frag_b(krow, 32, hi);
                v8f sv = {};
                sv = WMMA16(aq0, bk0, sv);
                sv = WMMA16(aq1, bk1, sv);
                int j = kt * 16 + lo16;
#pragma unroll
                for (int r = 0; r < 8; ++r) {
                    int i = q0 + r + 8 * hi;
                    float x = ((j <= i) && (j >= i - (WW - 1))) ? sv[r] : MASK_NEG;
                    float rm = x;
                    rm = fmaxf(rm, __shfl_xor(rm, 1, 32));
                    rm = fmaxf(rm, __shfl_xor(rm, 2, 32));
                    rm = fmaxf(rm, __shfl_xor(rm, 4, 32));
                    rm = fmaxf(rm, __shfl_xor(rm, 8, 32));
                    float mnew = fmaxf(mrow[r], rm);
                    float sc = __expf(mrow[r] - mnew);
                    float p  = __expf(x - mnew);
                    float rs = p;
                    rs += __shfl_xor(rs, 1, 32);
                    rs += __shfl_xor(rs, 2, 32);
                    rs += __shfl_xor(rs, 4, 32);
                    rs += __shfl_xor(rs, 8, 32);
                    lrow[r] = lrow[r] * sc + rs;
                    mrow[r] = mnew;
#pragma unroll
                    for (int dt = 0; dt < 4; ++dt) o[dt][r] *= sc;
                    st[(r + 8 * hi) * 32 + s * 16 + lo16] = (_Float16)p;
                }
            } else {
#pragma unroll
                for (int r = 0; r < 8; ++r)
                    st[(r + 8 * hi) * 32 + 16 + lo16] = (_Float16)0.f;
            }
        }
        asm volatile("s_wait_dscnt 0" ::: "memory");   // P staged -> readable
        v16h pa = frag_a(st + lo16 * 32, 0, hi);       // A-layout 16x32 from LDS
#pragma unroll
        for (int dt = 0; dt < 4; ++dt) {
            const _Float16* vrow = Vp + (size_t)(dt * 16 + lo16) * TT + kt2 * 16;
            int tb = kt2 * 16 + hi * 16;               // clamp at sequence tail
            v16h bv;
            if (tb < TT) {
                bv = frag_b(vrow, 0, hi);
            } else {
#pragma unroll
                for (int i = 0; i < 16; ++i) bv[i] = (_Float16)0.f;
            }
            o[dt] = WMMA16(pa, bv, o[dt]);
        }
    }

    const int b = bh >> 4, h = bh & 15;
#pragma unroll
    for (int r = 0; r < 8; ++r) {
        float inv = 1.0f / lrow[r];
        int m = b * TT + q0 + r + 8 * hi;
#pragma unroll
        for (int dt = 0; dt < 4; ++dt) {
            int col = h * DD + dt * 16 + lo16;
            Ao[(size_t)m * EE + col] = (_Float16)(o[dt][r] * inv);
        }
    }
}

// ---------------------------------------------------------------------------
// Output GEMM: Y[4096,1024] = Ao[4096,1024] @ Wout^T, fp32 result.
// ---------------------------------------------------------------------------
__global__ __launch_bounds__(128) void out_gemm(const _Float16* __restrict__ A,
                                                const _Float16* __restrict__ Wo,
                                                float* __restrict__ Y) {
    const int lane = threadIdx.x & 31;
    const int wave = threadIdx.x >> 5;
    const int hi   = lane >> 4;
    const int lo16 = lane & 15;
    const int m0 = blockIdx.x * 64 + wave * 16;
    const int n0 = blockIdx.y * 64;

    const _Float16* arow = A + (size_t)(m0 + lo16) * EE;
    v8f acc[4] = {{}, {}, {}, {}};
    gemm_core(arow,
              Wo + (size_t)(n0 +  0 + lo16) * EE,
              Wo + (size_t)(n0 + 16 + lo16) * EE,
              Wo + (size_t)(n0 + 32 + lo16) * EE,
              Wo + (size_t)(n0 + 48 + lo16) * EE, hi, acc);

#pragma unroll
    for (int nt = 0; nt < 4; ++nt) {
        int n = n0 + nt * 16 + lo16;
#pragma unroll
        for (int r = 0; r < 8; ++r) {
            int m = m0 + r + 8 * hi;
            Y[(size_t)m * EE + n] = acc[nt][r];
        }
    }
}

// ---------------------------------------------------------------------------
extern "C" void kernel_launch(void* const* d_in, const int* in_sizes, int n_in,
                              void* d_out, int out_size, void* d_ws, size_t ws_size,
                              hipStream_t stream) {
    (void)in_sizes; (void)n_in; (void)out_size; (void)ws_size;
    const float* x     = (const float*)d_in[0];   // [B,T,E]
    const float* wqkv  = (const float*)d_in[1];   // [3E,E]
    const float* wout  = (const float*)d_in[2];   // [E,E]
    float* out = (float*)d_out;                   // [B,T,E]

    _Float16* ws = (_Float16*)d_ws;
    size_t off = 0;
    _Float16* xh  = ws + off; off += (size_t)BB * TT * EE;        // 4M
    _Float16* wqh = ws + off; off += (size_t)3 * EE * EE;         // 3M
    _Float16* woh = ws + off; off += (size_t)EE * EE;             // 1M
    _Float16* Qh  = ws + off; off += (size_t)BB * HH * TT * DD;   // 4M
    _Float16* Kh  = ws + off; off += (size_t)BB * HH * TT * DD;   // 4M
    _Float16* Vt  = ws + off; off += (size_t)BB * HH * DD * TT;   // 4M
    _Float16* Ao  = ws + off;                                     // 4M

    const int n_x = BB * TT * EE, n_wq = 3 * EE * EE, n_wo = EE * EE;
    cvt_f32_f16<<<n_x  / 1024, 256, 0, stream>>>(x,    xh,  n_x);
    cvt_f32_f16<<<n_wq / 1024, 256, 0, stream>>>(wqkv, wqh, n_wq);
    cvt_f32_f16<<<n_wo / 1024, 256, 0, stream>>>(wout, woh, n_wo);

    qkv_gemm<<<dim3((BB * TT) / 64, (3 * EE) / 64), 128, 0, stream>>>(xh, wqh, Qh, Kh, Vt);
    attn_kernel<<<dim3((TT / 16) / 4, BB * HH), 128, 0, stream>>>(Qh, Kh, Vt, Ao);
    out_gemm<<<dim3((BB * TT) / 64, EE / 64), 128, 0, stream>>>(Ao, woh, out);
}
